// FlashAttention_25881472926067
// MI455X (gfx1250) — matrix operand order repaired and checked by reading, not compiled, not run on hardware
//
#include <hip/hip_runtime.h>

// FlashAttention forward, CDNA5 (gfx1250), wave32, f16 WMMA with f32 accumulate.
// One wave (32 threads) processes a 16-row Q tile for one (b, h).
// Scores are computed TRANSPOSED (S^T = K * Q^T) so that:
//   - softmax reduction over s is 8 in-lane values + one shfl_xor(16)
//   - exp(S - m) probabilities are already in the WMMA A-operand layout
//     for the P*V matmul (no LDS memory traffic, no cross-lane transpose).
// The s-loop is split: interior chunks skip causal compares entirely; only
// the diagonal chunk(s) run the per-element causal mask. K rows 0..15 and V rows 16..31 of the next
// chunk are prefetched (global_prefetch_b8) while current WMMAs execute.

typedef __attribute__((ext_vector_type(16))) _Float16 v16h;
typedef __attribute__((ext_vector_type(8)))  float    v8f;

#define HEADS 8
#define DIM   64
#define TLEN  2048
#define SEQ   2048
#define BQ    16

#define WMMA_F16(A, B, C) \
  __builtin_amdgcn_wmma_f32_16x16x32_f16(false, (A), false, (B), (short)0, (C), false, false)

// 16 consecutive floats -> 16 packed f16 (B-operand half-run)
__device__ __forceinline__ v16h load_run16(const float* __restrict__ p) {
  v16h r;
#pragma unroll
  for (int i = 0; i < 16; i += 4) {
    const float4 f = *reinterpret_cast<const float4*>(p + i);
    r[i + 0] = (_Float16)f.x; r[i + 1] = (_Float16)f.y;
    r[i + 2] = (_Float16)f.z; r[i + 3] = (_Float16)f.w;
  }
  return r;
}

// A-operand 16x32 f16 chunk for this lane: two runs of 8 consecutive floats
// (K = 8*half + 0..7 and K = 16 + 8*half + 0..7 within the 32-wide chunk).
__device__ __forceinline__ v16h load_a16(const float* __restrict__ row, int d0, int d1) {
  v16h r;
#pragma unroll
  for (int i = 0; i < 8; i += 4) {
    const float4 f = *reinterpret_cast<const float4*>(row + d0 + i);
    r[i + 0] = (_Float16)f.x; r[i + 1] = (_Float16)f.y;
    r[i + 2] = (_Float16)f.z; r[i + 3] = (_Float16)f.w;
  }
#pragma unroll
  for (int i = 0; i < 8; i += 4) {
    const float4 f = *reinterpret_cast<const float4*>(row + d1 + i);
    r[8 + i + 0] = (_Float16)f.x; r[8 + i + 1] = (_Float16)f.y;
    r[8 + i + 2] = (_Float16)f.z; r[8 + i + 3] = (_Float16)f.w;
  }
  return r;
}

__global__ __launch_bounds__(32)
void fa_fwd_wmma(const float* __restrict__ q,           // (B, T, H, D)
                 const float* __restrict__ kv,          // (B, S, 2, H, D)
                 const unsigned char* __restrict__ kpm, // (B, S) bool
                 const int* __restrict__ causal_p,
                 float* __restrict__ out)               // (B, T, H, D)
{
  const int lane = threadIdx.x;      // 0..31
  const int ln   = lane & 15;        // N index / softmax column
  const int half = lane >> 4;        // lane half
  const int tq0  = blockIdx.x * BQ;  // first Q row of this tile
  const int h    = blockIdx.y;
  const int b    = blockIdx.z;
  const int causal = causal_p[0];
  const int tg   = tq0 + ln;         // this lane's softmax column (global t)
  const float kscale = 0.125f;       // D^-0.5 = 64^-0.5
  const float NEG_INF = -__builtin_inff();

  // ---- Q as B-operand of K*Q^T: lane holds Q[t = tq0+ln][d], d in two
  //      8-wide runs per 32-chunk (K = 8*half + 0..7 and 16 + 8*half + 0..7).
  const float* qrow = q + (((size_t)(b * TLEN + tg)) * HEADS + h) * DIM;
  const v16h qb0 = load_a16(qrow, 8 * half, 16 + 8 * half);        // D chunk 0..31
  const v16h qb1 = load_a16(qrow, 32 + 8 * half, 48 + 8 * half);   // D chunk 32..63

  v8f acc[4];                         // out tiles (t x 16d), 4 tiles cover D=64
#pragma unroll
  for (int nt = 0; nt < 4; ++nt) acc[nt] = (v8f){};

  float m = NEG_INF;                  // running max for column t = tg
  float l = 0.0f;                     // running denominator

  // Loop bounds: total extent, and extent of causally-unmasked interior.
  int sEnd = causal ? ((tq0 + BQ + 31) & ~31) : SEQ;
  if (sEnd > SEQ) sEnd = SEQ;
  // Chunk [sc, sc+31] needs causal masking iff sc+31 > tq0 (min row of tile).
  int sFull = causal ? ((tq0 >= 31) ? ((((tq0 - 31) >> 5) + 1) << 5) : 0) : sEnd;

  const size_t vstride = (size_t)2 * HEADS * DIM;       // s-stride in kv (floats)
  const unsigned char* kpmb = kpm + (size_t)b * SEQ;

  auto process_chunk = [&](int sc, bool causalMask) {
    // Prefetch next chunk's K row (sc+32+ln) and V row (sc+48+ln) for this lane while WMMAs run.
    if (sc + 32 < sEnd) {
      __builtin_prefetch(
          kv + ((((size_t)(b * SEQ + sc + 32 + ln)) * 2 + 0) * HEADS + h) * DIM, 0, 0);
      __builtin_prefetch(
          kv + ((((size_t)(b * SEQ + sc + 32 + 16 + ln)) * 2 + 1) * HEADS + h) * DIM, 0, 0);
    }

    // ---- S^T tiles: A = K rows (16 s x 32 d), B = Q^T. C layout holds
    //      (row s = sc + 16n + r + 8*half, col t = tq0 + ln) in st{n}[r].
    v8f st0 = (v8f){}, st1 = (v8f){};
    {
      const float* krow = kv + ((((size_t)(b * SEQ + sc + ln)) * 2 + 0) * HEADS + h) * DIM;
      const v16h ka0 = load_a16(krow, 8 * half, 16 + 8 * half);
      const v16h ka1 = load_a16(krow, 32 + 8 * half, 48 + 8 * half);
      st0 = WMMA_F16(ka0, qb0, st0);
      st0 = WMMA_F16(ka1, qb1, st0);
    }
    {
      const float* krow = kv + ((((size_t)(b * SEQ + sc + 16 + ln)) * 2 + 0) * HEADS + h) * DIM;
      const v16h ka0 = load_a16(krow, 8 * half, 16 + 8 * half);
      const v16h ka1 = load_a16(krow, 32 + 8 * half, 48 + 8 * half);
      st1 = WMMA_F16(ka0, qb0, st1);
      st1 = WMMA_F16(ka1, qb1, st1);
    }

    // ---- scale + padding mask (+ causal mask only on diagonal chunks) ----
#pragma unroll
    for (int r = 0; r < 8; ++r) {
      const int s0 = sc + 8 * half + r;        // tile n=0 row
      const int s1 = s0 + 16;                  // tile n=1 row
      bool bad0 = !kpmb[s0];
      bool bad1 = !kpmb[s1];
      if (causalMask) {                        // compile-time folded per call site
        bad0 = bad0 || (s0 > tg);
        bad1 = bad1 || (s1 > tg);
      }
      st0[r] = bad0 ? NEG_INF : st0[r] * kscale;
      st1[r] = bad1 ? NEG_INF : st1[r] * kscale;
    }

    // ---- online softmax for column t = tg (16 values: 8 regs x 2 halves) ----
    float mt = st0[0];
#pragma unroll
    for (int r = 0; r < 8; ++r) { mt = fmaxf(mt, st0[r]); mt = fmaxf(mt, st1[r]); }
    mt = fmaxf(mt, __shfl_xor(mt, 16));
    const float mn     = fmaxf(m, mt);
    const float escale = __expf(m - mn);       // rescale factor for old acc / l

    float ls = 0.0f;
    v16h pa;                                   // P chunk already in A layout
#pragma unroll
    for (int r = 0; r < 8; ++r) {
      const float p0 = __expf(st0[r] - mn);
      const float p1 = __expf(st1[r] - mn);
      ls += p0 + p1;
      pa[r]     = (_Float16)p0;                // A pairs v=0..3 : K = 8*half + r
      pa[8 + r] = (_Float16)p1;                // A pairs v=4..7 : K = 16 + 8*half + r
    }
    ls += __shfl_xor(ls, 16);
    l = l * escale + ls;
    m = mn;

    // ---- rescale output accumulators: factor for out-row (r + 8*half)
    //      lives in lane (r + 8*half) -> index shuffle (ds_bpermute) ----
#pragma unroll
    for (int r = 0; r < 8; ++r) {
      const float fr = __shfl(escale, 8 * half + r);
      acc[0][r] *= fr; acc[1][r] *= fr; acc[2][r] *= fr; acc[3][r] *= fr;
    }

    // ---- P * V : B operand lane holds V[s = sc + 8*half + e + (e & 8)][d = 16*nt + ln] ----
    const float* vbase =
        kv + ((((size_t)(b * SEQ + sc + 8 * half)) * 2 + 1) * HEADS + h) * DIM + ln;
#pragma unroll
    for (int nt = 0; nt < 4; ++nt) {
      v16h vb;
#pragma unroll
      for (int e = 0; e < 16; ++e)
        vb[e] = (_Float16)vbase[(size_t)(e + (e & 8)) * vstride + 16 * nt];
      acc[nt] = WMMA_F16(pa, vb, acc[nt]);
    }
  };

  int sc = 0;
  for (; sc < sFull; sc += 32) process_chunk(sc, false);  // no causal compares
  for (; sc < sEnd;  sc += 32) process_chunk(sc, true);   // diagonal chunk(s)

  // ---- normalize and store: out row t = tq0 + r + 8*half, col d = 16*nt + ln ----
#pragma unroll
  for (int r = 0; r < 8; ++r) {
    const float lr  = __shfl(l, 8 * half + r);
    const float inv = 1.0f / lr;
    const int   t   = tq0 + 8 * half + r;
    float* orow = out + (((size_t)(b * TLEN + t)) * HEADS + h) * DIM + ln;
    orow[0]  = acc[0][r] * inv;
    orow[16] = acc[1][r] * inv;
    orow[32] = acc[2][r] * inv;
    orow[48] = acc[3][r] * inv;
  }
}

extern "C" void kernel_launch(void* const* d_in, const int* in_sizes, int n_in,
                              void* d_out, int out_size, void* d_ws, size_t ws_size,
                              hipStream_t stream) {
  const float*         qp   = (const float*)d_in[0];
  const float*         kvp  = (const float*)d_in[1];
  const unsigned char* kpm  = (const unsigned char*)d_in[2];
  const int*           caus = (const int*)d_in[3];
  float*               outp = (float*)d_out;

  const int B = in_sizes[0] / (TLEN * HEADS * DIM);   // = 2 for this problem
  dim3 grid(TLEN / BQ, HEADS, B);                     // 128 x 8 x 2 waves
  fa_fwd_wmma<<<grid, 32, 0, stream>>>(qp, kvp, kpm, caus, outp);
}
